// ReadUnit_9371618640097
// MI455X (gfx1250) — compile-verified, matrix operand order checked by reading, not run on hardware
//
#include <hip/hip_runtime.h>
#include <hip/hip_bf16.h>

typedef __attribute__((ext_vector_type(16))) __bf16       v16bf;
typedef __attribute__((ext_vector_type(8)))  float        v8f;
typedef __attribute__((ext_vector_type(8)))  unsigned int v8u;

union BF16Frag {
    v8u            v;   // 8 dwords = 32 bytes = 16 bf16
    v16bf          b;
    unsigned short u[16];
};

__device__ __forceinline__ unsigned short f32_to_bf16(float f) {
    unsigned int x = __float_as_uint(f);
    x += 0x7FFFu + ((x >> 16) & 1u);   // round-to-nearest-even
    return (unsigned short)(x >> 16);
}

__device__ __forceinline__ float bf16_to_f32(unsigned short h) {
    return __uint_as_float(((unsigned int)h) << 16);
}

#define D        512
#define NROW     112            // rows (tokens) per workgroup tile; 2 tiles cover 196
#define LDSK     1056           // LDS row stride in bytes (512*2 + 32 pad vs bank aliasing)
#define MT       7              // NROW/16 M-tiles
#define NT       32             // D/16 N-tiles
#define KSTEPS   16             // D/32 k-steps per WMMA chain

// ---------------------------------------------------------------------------
// Weight repack: f32 [k][n] -> bf16 WMMA B-fragments.
// Fragment (kblock, ntile) = 32(K) x 16(N) bf16 = 1024 B; lane L owns bytes
// [L*32, L*32+32): col = L%16, K = (L/16)*16 + e.
// Packed order: [w:4][kblock:16][ntile:32][lane:32][e:16]
// w: 0=W_m, 1=W_kb, 2=W_merge[:512], 3=W_merge[512:]
// ---------------------------------------------------------------------------
__global__ __launch_bounds__(256) void repack_weights(
    const float* __restrict__ Wm, const float* __restrict__ Wkb,
    const float* __restrict__ Wmerge, unsigned short* __restrict__ out)
{
    int idx = blockIdx.x * 256 + threadIdx.x;      // 0 .. 4*262144-1
    int w      = idx >> 18;
    int r      = idx & 262143;
    int frag   = r >> 9;                           // kblock*32 + ntile
    int within = r & 511;
    int lane   = within >> 4;
    int e      = within & 15;
    int ntile  = frag & 31;
    int kblock = frag >> 5;
    int k = kblock * 32 + (lane >> 4) * 16 + e;
    int n = ntile * 16 + (lane & 15);
    float v;
    if      (w == 0) v = Wm[k * D + n];
    else if (w == 1) v = Wkb[k * D + n];
    else if (w == 2) v = Wmerge[k * D + n];
    else             v = Wmerge[(D + k) * D + n];
    out[idx] = f32_to_bf16(v);
}

__global__ __launch_bounds__(256) void zero_out(float* __restrict__ out) {
    out[blockIdx.x * 256 + threadIdx.x] = 0.0f;
}

// ---------------------------------------------------------------------------
// mproj = m_prev @ W_m + b_m     [128,512] x [512,512]
// One wave per 16x16 output tile; 8 waves/block, 32 blocks.
// ---------------------------------------------------------------------------
__global__ __launch_bounds__(256) void mprev_proj_kernel(
    const float* __restrict__ m_prev, const float* __restrict__ b_m,
    const unsigned short* __restrict__ wm_packed, float* __restrict__ mproj)
{
    const int lane = threadIdx.x & 31;
    const int wid  = threadIdx.x >> 5;
    const int tile = blockIdx.x * 8 + wid;     // 256 tiles
    const int mt   = tile & 7;                 // 8 M-tiles (128 rows)
    const int nt   = tile >> 3;                // 32 N-tiles

    const int arow  = mt * 16 + (lane & 15);
    const int kbase = (lane >> 4) * 16;

    v8f acc = {};
    for (int kb = 0; kb < KSTEPS; ++kb) {
        BF16Frag a;
        const float* ap = m_prev + arow * D + kb * 32 + kbase;
        #pragma unroll
        for (int e = 0; e < 16; ++e) a.u[e] = f32_to_bf16(ap[e]);
        BF16Frag b;
        b.v = *(const v8u*)(wm_packed + (kb * 32 + nt) * 512 + lane * 16);
        acc = __builtin_amdgcn_wmma_f32_16x16x32_bf16(
            false, a.b, false, b.v, (short)0, acc, false, false);
    }
    const int col  = nt * 16 + (lane & 15);
    const float bias = b_m[col];
    const int r0 = mt * 16 + 8 * (lane >> 4);
    #pragma unroll
    for (int v = 0; v < 8; ++v) mproj[(r0 + v) * D + col] = acc[v] + bias;
}

// ---------------------------------------------------------------------------
// Fused main kernel: one workgroup = (batch, row-block of 112 tokens).
// nt-outer / mt-inner loop order keeps each column tile's B fragments in
// registers across all 7 M-tiles (7x less L2 weight traffic).
// ---------------------------------------------------------------------------
__global__ __launch_bounds__(256) void fused_kernel(
    const float* __restrict__ KB,      // [B,512,14,14]
    const float* __restrict__ c_i,     // [B,512]
    const float* __restrict__ mproj,   // [B,512]
    const float* __restrict__ b_kb,    // [512]
    const float* __restrict__ b_merge, // [512]
    const float* __restrict__ W_attn,  // [512]
    const float* __restrict__ b_attn,  // [1]
    const unsigned short* __restrict__ wkb_p,
    const unsigned short* __restrict__ wm1_p,
    const unsigned short* __restrict__ wm2_p,
    float* __restrict__ out)           // [B,512], pre-zeroed
{
    extern __shared__ char smem[];
    char*  Xb  = smem;                              // [NROW][LDSK] bf16 X tile
    char*  Ib  = smem + NROW * LDSK;                // [NROW][LDSK] bf16 I tile
    float* mv  = (float*)(smem + 2 * NROW * LDSK);  // [NROW]

    const int tid  = threadIdx.x;
    const int lane = tid & 31;
    const int wid  = tid >> 5;
    const int b    = blockIdx.x >> 1;
    const int row0 = (blockIdx.x & 1) * NROW;

    if (tid < NROW) mv[tid] = 0.0f;

    // ---- Load X = KB_hw[b, row0:row0+112, :] into LDS as bf16 ----
    // KB_hw[b, t, c] = KB[(b*512 + c)*196 + t]; consecutive tid -> consecutive t.
    const float* kbbase = KB + (size_t)b * D * 196;
    for (int i = tid; i < NROW * D; i += 256) {
        int c = i / NROW;
        int r = i - c * NROW;
        int t = row0 + r; if (t > 195) t = 195;     // clamp pad rows (masked later)
        *(unsigned short*)(Xb + r * LDSK + c * 2) = f32_to_bf16(kbbase[c * 196 + t]);
    }
    __syncthreads();

    const int acol  = lane & 15;
    const int kbase = (lane >> 4) * 16;
    const v8f zacc = {};

    // ---- Stage 1: I = mproj[b] * (X @ W_kb + b_kb), bf16 into LDS ----
    for (int nt = wid; nt < NT; nt += 8) {          // 4 column tiles per wave
        BF16Frag wf[KSTEPS];
        #pragma unroll
        for (int kb = 0; kb < KSTEPS; ++kb)
            wf[kb].v = *(const v8u*)(wkb_p + (kb * 32 + nt) * 512 + lane * 16);

        const int col   = nt * 16 + acol;
        const float bias = b_kb[col];
        const float mp   = mproj[b * D + col];

        #pragma unroll
        for (int mt = 0; mt < MT; ++mt) {
            const int arow = mt * 16 + acol;
            v8f acc = zacc;
            #pragma unroll
            for (int kb = 0; kb < KSTEPS; ++kb) {
                BF16Frag a;
                a.v = *(const v8u*)(Xb + arow * LDSK + (kb * 32 + kbase) * 2);
                acc = __builtin_amdgcn_wmma_f32_16x16x32_bf16(
                    false, a.b, false, wf[kb].b, (short)0, acc, false, false);
            }
            const int r0 = mt * 16 + 8 * (lane >> 4);
            #pragma unroll
            for (int v = 0; v < 8; ++v) {
                float val = (acc[v] + bias) * mp;
                *(unsigned short*)(Ib + (r0 + v) * LDSK + col * 2) = f32_to_bf16(val);
            }
        }
    }
    __syncthreads();

    // ---- Stage 2: _I = I@Wm1 + X@Wm2 + b_merge; I2 = c_i*_I; mv += I2@W_attn ----
    for (int nt = wid; nt < NT; nt += 8) {
        v8f accs[MT];
        #pragma unroll
        for (int mt = 0; mt < MT; ++mt) accs[mt] = zacc;

        // K split in two halves so B fragments (8+8) + 7 accumulators fit in VGPRs
        #pragma unroll
        for (int kh = 0; kh < 2; ++kh) {
            BF16Frag w1f[8], w2f[8];
            #pragma unroll
            for (int k8 = 0; k8 < 8; ++k8) {
                const int kb = kh * 8 + k8;
                w1f[k8].v = *(const v8u*)(wm1_p + (kb * 32 + nt) * 512 + lane * 16);
                w2f[k8].v = *(const v8u*)(wm2_p + (kb * 32 + nt) * 512 + lane * 16);
            }
            #pragma unroll
            for (int mt = 0; mt < MT; ++mt) {
                const int arow = mt * 16 + acol;
                v8f acc = accs[mt];
                #pragma unroll
                for (int k8 = 0; k8 < 8; ++k8) {
                    const int koff = ((kh * 8 + k8) * 32 + kbase) * 2;
                    BF16Frag ai, ax;
                    ai.v = *(const v8u*)(Ib + arow * LDSK + koff);
                    acc = __builtin_amdgcn_wmma_f32_16x16x32_bf16(
                        false, ai.b, false, w1f[k8].b, (short)0, acc, false, false);
                    ax.v = *(const v8u*)(Xb + arow * LDSK + koff);
                    acc = __builtin_amdgcn_wmma_f32_16x16x32_bf16(
                        false, ax.b, false, w2f[k8].b, (short)0, acc, false, false);
                }
                accs[mt] = acc;
            }
        }

        const int col = nt * 16 + acol;
        const float bias = b_merge[col];
        const float ci   = c_i[b * D + col];
        const float wa   = W_attn[col];
        #pragma unroll
        for (int mt = 0; mt < MT; ++mt) {
            const int r0 = mt * 16 + 8 * (lane >> 4);
            float p[8];
            #pragma unroll
            for (int v = 0; v < 8; ++v) p[v] = (accs[mt][v] + bias) * ci * wa;
            // reduce over the 16 columns of this tile (lanes 0-15 / 16-31 groups)
            #pragma unroll
            for (int off = 1; off < 16; off <<= 1) {
                #pragma unroll
                for (int v = 0; v < 8; ++v) p[v] += __shfl_xor(p[v], off, 32);
            }
            if ((lane & 15) == 0) {
                #pragma unroll
                for (int v = 0; v < 8; ++v) atomicAdd(&mv[r0 + v], p[v]);
            }
        }
    }
    __syncthreads();

    // finalize mv: add b_attn, mask padded rows
    const float ba = b_attn[0];
    if (tid < NROW) {
        int t = row0 + tid;
        mv[tid] = (t < 196) ? (mv[tid] + ba) : 0.0f;
    }
    __syncthreads();

    // ---- Stage 3: m_new[b, c] += sum_r mv[r] * X[r, c] ----
    for (int c = tid; c < D; c += 256) {
        float sum = 0.0f;
        for (int r = 0; r < NROW; ++r) {
            unsigned short xb = *(const unsigned short*)(Xb + r * LDSK + c * 2);
            sum += mv[r] * bf16_to_f32(xb);
        }
        atomicAdd(&out[b * D + c], sum);
    }
}

extern "C" void kernel_launch(void* const* d_in, const int* in_sizes, int n_in,
                              void* d_out, int out_size, void* d_ws, size_t ws_size,
                              hipStream_t stream) {
    const float* m_prev  = (const float*)d_in[0];
    const float* KB      = (const float*)d_in[1];
    const float* c_i     = (const float*)d_in[2];
    const float* W_m     = (const float*)d_in[3];
    const float* b_m     = (const float*)d_in[4];
    const float* W_kb    = (const float*)d_in[5];
    const float* b_kb    = (const float*)d_in[6];
    const float* W_merge = (const float*)d_in[7];
    const float* b_merge = (const float*)d_in[8];
    const float* W_attn  = (const float*)d_in[9];
    const float* b_attn  = (const float*)d_in[10];
    float* out = (float*)d_out;

    // workspace: 2 MB packed bf16 weights + 256 KB mproj
    unsigned short* wpack = (unsigned short*)d_ws;
    const size_t PACK_PER_W = 512u * 512u;             // elements per weight
    unsigned short* wm_p  = wpack + 0 * PACK_PER_W;
    unsigned short* wkb_p = wpack + 1 * PACK_PER_W;
    unsigned short* wm1_p = wpack + 2 * PACK_PER_W;
    unsigned short* wm2_p = wpack + 3 * PACK_PER_W;
    float* mproj = (float*)((char*)d_ws + 4 * PACK_PER_W * sizeof(unsigned short));

    const int SMEM = 2 * NROW * LDSK + NROW * (int)sizeof(float); // 236,992 B
    (void)hipFuncSetAttribute((const void*)fused_kernel,
                              hipFuncAttributeMaxDynamicSharedMemorySize, SMEM);

    repack_weights<<<4096, 256, 0, stream>>>(W_m, W_kb, W_merge, wpack);
    zero_out<<<(128 * 512) / 256, 256, 0, stream>>>(out);
    mprev_proj_kernel<<<32, 256, 0, stream>>>(m_prev, b_m, wm_p, mproj);
    fused_kernel<<<128 * 2, 256, SMEM, stream>>>(
        KB, c_i, mproj, b_kb, b_merge, W_attn, b_attn,
        wkb_p, wm1_p, wm2_p, out);
}
